// backbone_13331578487458
// MI455X (gfx1250) — matrix operand order repaired and checked by reading, not compiled, not run on hardware
//
#include <hip/hip_runtime.h>

typedef unsigned short u16;
typedef unsigned int   u32;
typedef __attribute__((ext_vector_type(4)))  u32    u32x4;
typedef __attribute__((ext_vector_type(4)))  int    i32x4;
typedef __attribute__((ext_vector_type(8)))  int    i32x8;
typedef __attribute__((ext_vector_type(16))) __bf16 v16bf;
typedef __attribute__((ext_vector_type(8)))  float  v8f;

#if defined(__HIP_DEVICE_COMPILE__) && defined(__has_builtin)
#  if __has_builtin(__builtin_amdgcn_tensor_load_to_lds)
#    define USE_TDM 1
#  endif
#endif
#ifndef USE_TDM
#  define USE_TDM 0
#endif

__device__ __forceinline__ u16 f2bf(float f){
    u32 u = __float_as_uint(f);
    u32 r = u + 0x7FFFu + ((u >> 16) & 1u);   // round-to-nearest-even
    return (u16)(r >> 16);
}
__device__ __forceinline__ float bf2f(u16 h){
    return __uint_as_float(((u32)h) << 16);
}

// ---------------- elementwise fp32 -> bf16 ----------------
__global__ void k_f32_to_bf16(const float* __restrict__ in, u16* __restrict__ out, int n){
    int i = blockIdx.x * 256 + threadIdx.x;
    if (i < n) out[i] = f2bf(in[i]);
}

// --- weight convert + transpose: w [K][Cout] fp32 (HWIO flat) -> wt [Cout][K] bf16 ---
__global__ void k_wxpose(const float* __restrict__ w, u16* __restrict__ wt, int K, int lCout){
    int i = blockIdx.x * 256 + threadIdx.x;
    if (i < (K << lCout)){
        int kk = i >> lCout;
        int co = i & ((1 << lCout) - 1);
        wt[(size_t)co * K + kk] = f2bf(w[i]);
    }
}

// ------------- 3x3 stride-2 pad-1 maxpool, NHWC bf16 (all dims pow2 -> shifts) -------------
__global__ void k_maxpool(const u16* __restrict__ in, u16* __restrict__ out,
                          int lw, int lC){
    const int W  = 1 << lw;           // input H == W
    const int Wo = W >> 1;
    const int lwo = lw - 1;
    long long total = (long long)2 * Wo * Wo << lC;
    long long t = (long long)blockIdx.x * 256 + threadIdx.x;
    if (t >= total) return;
    int c  = (int)(t & ((1 << lC) - 1));
    long long r0 = t >> lC;
    int ow = (int)(r0 & (Wo - 1)); r0 >>= lwo;
    int oh = (int)(r0 & (Wo - 1));
    int n  = (int)(r0 >> lwo);
    float best = -3.4e38f;
    for (int dy = 0; dy < 3; ++dy){
        int ih = oh * 2 - 1 + dy;
        if ((unsigned)ih >= (unsigned)W) continue;
        for (int dx = 0; dx < 3; ++dx){
            int iw = ow * 2 - 1 + dx;
            if ((unsigned)iw >= (unsigned)W) continue;
            best = fmaxf(best, bf2f(in[(((((size_t)(n << lw) + ih) << lw) + iw) << lC) + c]));
        }
    }
    out[t] = f2bf(best);
}

// ---------------- implicit-GEMM conv, bf16 WMMA, fused BN/residual/ReLU ----------------
// Macro tile 128(M) x 64(N); 8 waves each computing 32x32 (4 accumulators).
// K staged in 64-element chunks; B tile fetched by the Tensor Data Mover when available.
union FragU { u32x4 q[2]; v16bf v; };

template<int KDIM, int CIN>
__global__ __launch_bounds__(256)
void k_conv_wmma(const u16* __restrict__ in, const u16* __restrict__ wt,
                 const float* __restrict__ bias, const float* __restrict__ gamma,
                 const float* __restrict__ beta, const u16* __restrict__ res,
                 u16* __restrict__ outb, float* __restrict__ outf,
                 int lw, int Cout, int relu)
{
    constexpr int PAD = KDIM / 2;
    constexpr int K   = KDIM * KDIM * CIN;
    constexpr int NST = (K + 63) / 64;          // stages (tail zero-filled)

    __shared__ __align__(16) u16 As[128 * 64];  // 16 KB: [pixel-row][64 K]
    __shared__ __align__(16) u16 Bs[64 * 64];   //  8 KB: [co-col]   [64 K]

    const int W      = 1 << lw;                 // H == W, power of two
    const int m_base = blockIdx.x * 128;
    const int n_base = blockIdx.y * 64;
    const int tid    = threadIdx.x;

    // --- A-loader coords: one thread = one pixel row, one 32-elem K half (64B) ---
    const int arow  = tid >> 1;
    const int ahalf = tid & 1;
    const int am    = m_base + arow;
    const int animg = am >> (2 * lw);
    const int arem  = am & ((1 << (2 * lw)) - 1);
    const int aoh   = arem >> lw;
    const int aow   = arem & (W - 1);

    const int lane = tid & 31;
    const int hi   = lane >> 4;
    const int lrow = lane & 15;
    const int wave = tid >> 5;
    const int m0w  = (wave & 3) * 32;           // 4 waves along M
    const int n0w  = (wave >> 2) * 32;          // 2 waves along N

#if !USE_TDM
    // --- fallback B-loader coords: one thread = one co column, one 16-elem K seg ---
    const int bcol = tid & 63;
    const int bseg = tid >> 6;
    const int coB  = n_base + bcol;
#endif

    v8f acc[2][2];
    #pragma unroll
    for (int i = 0; i < 2; ++i)
        #pragma unroll
        for (int j = 0; j < 2; ++j)
            acc[i][j] = (v8f){0.f,0.f,0.f,0.f,0.f,0.f,0.f,0.f};

    for (int st = 0; st < NST; ++st){
        const int kk0 = st * 64;

        // ---- stage A tile (implicit SAME padding; single tap per 32-elem chunk) ----
        {
            const int kkc = kk0 + ahalf * 32;
            const int rs  = kkc / CIN;          // compile-time strength-reduced
            const int ci0 = kkc % CIN;
            const int kh  = rs / KDIM;
            const int kw_ = rs % KDIM;
            const int ih  = aoh + kh - PAD;
            const int iw  = aow + kw_ - PAD;
            u32x4 v0 = {0u,0u,0u,0u}, v1 = {0u,0u,0u,0u},
                  v2 = {0u,0u,0u,0u}, v3 = {0u,0u,0u,0u};
            if (kkc < K && (unsigned)ih < (unsigned)W && (unsigned)iw < (unsigned)W){
                const u16* pe = in + ((((size_t)(animg << lw) + ih) << lw) + iw) * CIN + ci0;
                const u32x4* p = (const u32x4*)pe;
                v0 = p[0]; v1 = p[1]; v2 = p[2]; v3 = p[3];
                if (kkc + 64 < K) __builtin_prefetch(pe + 64, 0, 0);  // global_prefetch_b8
            }
            u32x4* s = (u32x4*)&As[arow * 64 + ahalf * 32];
            s[0] = v0; s[1] = v1; s[2] = v2; s[3] = v3;
        }

#if USE_TDM
        // ---- stage B tile via Tensor Data Mover: 64x64 bf16 tile of wt[Cout][K] ----
        if (wave == 0){
            const u32 ldsB = (u32)(uintptr_t)(&Bs[0]);                  // LDS byte address
            const unsigned long long ga =
                (unsigned long long)(uintptr_t)(wt + (size_t)n_base * K + kk0);
            const u32 td0 = (u32)(K - kk0);        // remaining row length (elems)
            const u32 td1 = (u32)(Cout - n_base);  // remaining rows -> OOB rows read 0
            // D# group 0: count=1 | lds_addr | global_addr[56:0] | type=2
            u32x4 g0 = { 1u, ldsB, (u32)ga,
                         (u32)((ga >> 32) & 0x01FFFFFFull) | 0x80000000u };
            // D# group 1: data_size=2B; tensor_dim0/1; tile 64x64; dim0 stride = K
            i32x8 g1 = { (int)0x00010000,
                         (int)((td0 & 0xFFFFu) << 16),
                         (int)(((td0 >> 16) & 0xFFFFu) | ((td1 & 0xFFFFu) << 16)),
                         (int)(((td1 >> 16) & 0xFFFFu) | (64u << 16)),
                         (int)64,
                         (int)(u32)K,
                         0, 0 };
            i32x4 z4 = {0, 0, 0, 0};
#  if __clang_major__ >= 23
            i32x8 z8 = {0, 0, 0, 0, 0, 0, 0, 0};
            __builtin_amdgcn_tensor_load_to_lds(g0, g1, z4, z4, z8, 0);
#  else
            __builtin_amdgcn_tensor_load_to_lds(g0, g1, z4, z4, 0);
#  endif
            __builtin_amdgcn_s_wait_tensorcnt(0);
        }
#else
        // ---- fallback: cooperative B tile load (pre-transposed weights [Cout][K]) ----
        {
            const int kks = kk0 + bseg * 16;
            u32x4 v0 = {0u,0u,0u,0u}, v1 = {0u,0u,0u,0u};
            if (coB < Cout && kks < K){
                const u16* p = wt + (size_t)coB * K + kks;
                v0 = ((const u32x4*)p)[0];
                v1 = ((const u32x4*)p)[1];
                if (st + 1 < NST) __builtin_prefetch(p + 64, 0, 0);
            }
            u32x4* s = (u32x4*)&Bs[bcol * 64 + bseg * 16];
            s[0] = v0; s[1] = v1;
        }
#endif

        __syncthreads();

        #pragma unroll
        for (int ku = 0; ku < 2; ++ku){
            FragU a0, a1, b0, b1;
            const u16* ap0 = &As[(m0w + lrow) * 64 + ku * 32 + hi * 8];
            a0.q[0] = *(const u32x4*)ap0;
            a0.q[1] = *(const u32x4*)(ap0 + 16);
            const u16* ap1 = &As[(m0w + 16 + lrow) * 64 + ku * 32 + hi * 8];
            a1.q[0] = *(const u32x4*)ap1;
            a1.q[1] = *(const u32x4*)(ap1 + 16);
            const u16* bp0 = &Bs[(n0w + lrow) * 64 + ku * 32 + hi * 8];
            b0.q[0] = *(const u32x4*)bp0;
            b0.q[1] = *(const u32x4*)(bp0 + 16);
            const u16* bp1 = &Bs[(n0w + 16 + lrow) * 64 + ku * 32 + hi * 8];
            b1.q[0] = *(const u32x4*)bp1;
            b1.q[1] = *(const u32x4*)(bp1 + 16);

            acc[0][0] = __builtin_amdgcn_wmma_f32_16x16x32_bf16(false, a0.v, false, b0.v,
                                                               (short)0, acc[0][0], false, false);
            acc[0][1] = __builtin_amdgcn_wmma_f32_16x16x32_bf16(false, a0.v, false, b1.v,
                                                               (short)0, acc[0][1], false, false);
            acc[1][0] = __builtin_amdgcn_wmma_f32_16x16x32_bf16(false, a1.v, false, b0.v,
                                                               (short)0, acc[1][0], false, false);
            acc[1][1] = __builtin_amdgcn_wmma_f32_16x16x32_bf16(false, a1.v, false, b1.v,
                                                               (short)0, acc[1][1], false, false);
        }

        __syncthreads();
    }

    // ---- fused epilogue: bias, BN affine, residual, ReLU ----
    #pragma unroll
    for (int mi = 0; mi < 2; ++mi){
        #pragma unroll
        for (int ni = 0; ni < 2; ++ni){
            int co = n_base + n0w + ni * 16 + lrow;
            if (co >= Cout) continue;
            float bb = bias[co];
            float gg = gamma ? gamma[co] : 1.0f;
            float be = beta  ? beta[co]  : 0.0f;
            #pragma unroll
            for (int v = 0; v < 8; ++v){
                int mm = m_base + m0w + mi * 16 + v + hi * 8;  // D: lanes16-31 hold M=v+8
                float val = (acc[mi][ni][v] + bb) * gg + be;
                if (res)  val += bf2f(res[(size_t)mm * Cout + co]);
                if (relu) val = fmaxf(val, 0.0f);
                if (outf){                        // final conv: fp32 NCHW straight to d_out
                    int ni2 = mm >> (2 * lw);
                    int rm  = mm & ((1 << (2 * lw)) - 1);
                    int o_h = rm >> lw, o_w = rm & (W - 1);
                    outf[((((size_t)(ni2 * Cout + co) << lw) + o_h) << lw) + o_w] = val;
                } else {
                    outb[(size_t)mm * Cout + co] = f2bf(val);
                }
            }
        }
    }
}

// ===================== host orchestration =====================
struct ConvP {
    const u16* in; const u16* wt; const float* b; const float* g; const float* be;
    const u16* res; u16* outb; float* outf; int lw; int cout; int relu;
};

template<int KD, int CI>
static void launch_conv_t(const ConvP& p, hipStream_t stream){
    int M = 2 << (2 * p.lw);                    // N=2, H=W=1<<lw; multiple of 128
    dim3 grid(M / 128, (p.cout + 63) / 64);
    k_conv_wmma<KD, CI><<<grid, dim3(256), 0, stream>>>(
        p.in, p.wt, p.b, p.g, p.be, p.res, p.outb, p.outf, p.lw, p.cout, p.relu);
}

static void launch_conv(int k, int cin, const ConvP& p, hipStream_t s){
    if (k == 1){
        if      (cin == 32)  launch_conv_t<1, 32>(p, s);
        else if (cin == 64)  launch_conv_t<1, 64>(p, s);
        else if (cin == 128) launch_conv_t<1, 128>(p, s);
        else if (cin == 256) launch_conv_t<1, 256>(p, s);
        else                 launch_conv_t<1, 512>(p, s);
    }
    else if (k == 11) launch_conv_t<11, 32>(p, s);
    else if (k == 7)  launch_conv_t<7, 64>(p, s);
    else if (k == 5)  launch_conv_t<5, 128>(p, s);
    else              launch_conv_t<3, 512>(p, s);
}

extern "C" void kernel_launch(void* const* d_in, const int* in_sizes, int n_in,
                              void* d_out, int out_size, void* d_ws, size_t ws_size,
                              hipStream_t stream)
{
    (void)in_sizes; (void)n_in; (void)out_size; (void)ws_size;
    // jax pytree (sorted-key) flattening:
    //   blocks d1,d1_1,d2,d2_1,d3,d3_1 at base 0,16,..,80 with key order
    //   (b1,b2,b3,be1,be2,be3,bes,bs,g1,g2,g3,gs,w1,w2,w3,ws); d4.b=96, d4.w=97, x=98
    const float* x = (const float*)d_in[98];

    char* ws = (char*)d_ws;
    size_t off = 0;
    auto balloc = [&](size_t bytes) -> void* {
        void* p = ws + off;
        off = (off + bytes + 255) & ~(size_t)255;
        return p;
    };
    const size_t ABUF = (size_t)131072 * 128 * sizeof(u16);
    u16* act[4];
    for (int i = 0; i < 4; ++i) act[i] = (u16*)balloc(ABUF);

    struct Conv { const float* w; const float* b; const float* g; const float* be;
                  int k, cin, cout; u16* wt; };
    struct BlkCfg { int cin, cout, k; };
    const BlkCfg bc[6] = {{64,128,11},{128,128,11},{128,256,7},{256,256,7},{256,512,5},{512,512,5}};

    Conv cv[25];
    int nc = 0;
    for (int b = 0; b < 6; ++b){
        int base = b * 16;
        const float* bias1 = (const float*)d_in[base + 0];
        const float* bias2 = (const float*)d_in[base + 1];
        const float* bias3 = (const float*)d_in[base + 2];
        const float* be1   = (const float*)d_in[base + 3];
        const float* be2   = (const float*)d_in[base + 4];
        const float* be3   = (const float*)d_in[base + 5];
        const float* bes   = (const float*)d_in[base + 6];
        const float* biass = (const float*)d_in[base + 7];
        const float* g1    = (const float*)d_in[base + 8];
        const float* g2    = (const float*)d_in[base + 9];
        const float* g3    = (const float*)d_in[base + 10];
        const float* gs    = (const float*)d_in[base + 11];
        const float* w1    = (const float*)d_in[base + 12];
        const float* w2    = (const float*)d_in[base + 13];
        const float* w3    = (const float*)d_in[base + 14];
        const float* wsh   = (const float*)d_in[base + 15];
        int cin = bc[b].cin, cout = bc[b].cout, k = bc[b].k, c4 = cout / 4;
        cv[nc++] = {w1,  bias1, g1, be1, 1, cin, c4,   nullptr};
        cv[nc++] = {w2,  bias2, g2, be2, k, c4,  c4,   nullptr};
        cv[nc++] = {w3,  bias3, g3, be3, 1, c4,  cout, nullptr};
        cv[nc++] = {wsh, biass, gs, bes, 1, cin, cout, nullptr};
    }
    cv[nc++] = {(const float*)d_in[97], (const float*)d_in[96], nullptr, nullptr, 3, 512, 1024, nullptr};

    // convert + transpose all weights to bf16 [Cout][K] (Cout is a power of two)
    for (int i = 0; i < 25; ++i){
        int K = cv[i].k * cv[i].k * cv[i].cin;
        int lCout = 31 - __builtin_clz((unsigned)cv[i].cout);
        size_t elems = (size_t)K * cv[i].cout;
        cv[i].wt = (u16*)balloc(elems * sizeof(u16));
        int grid = (int)((elems + 255) / 256);
        k_wxpose<<<dim3(grid), dim3(256), 0, stream>>>(cv[i].w, cv[i].wt, K, lCout);
    }

    // convert input activations to bf16
    {
        int n = 2 * 256 * 256 * 64;
        k_f32_to_bf16<<<dim3((n + 255) / 256), dim3(256), 0, stream>>>(x, act[0], n);
    }

    auto conv = [&](int ci, const u16* in, const u16* resid, u16* outb, float* outf,
                    int lw, int relu){
        const Conv& c = cv[ci];
        ConvP p = {in, c.wt, c.b, c.g, c.be, resid, outb, outf, lw, c.cout, relu};
        launch_conv(c.k, c.cin, p, stream);
    };

    auto bottleneck = [&](int c0, const u16* X, u16* t1, u16* t2, u16* Y, int lw){
        conv(c0 + 0, X,  nullptr, t1, nullptr, lw, 1);   // 1x1 + BN + ReLU
        conv(c0 + 1, t1, nullptr, t2, nullptr, lw, 1);   // kxk + BN + ReLU
        conv(c0 + 3, X,  nullptr, t1, nullptr, lw, 0);   // shortcut 1x1 + BN
        conv(c0 + 2, t2, t1,      Y,  nullptr, lw, 1);   // 1x1 + BN + add + ReLU
    };

    auto pool = [&](const u16* in, u16* out, int lw, int C){
        int lC = 31 - __builtin_clz((unsigned)C);
        long long total = (long long)2 * (1 << (lw - 1)) * (1 << (lw - 1)) * C;
        k_maxpool<<<dim3((unsigned)((total + 255) / 256)), dim3(256), 0, stream>>>(
            in, out, lw, lC);
    };

    bottleneck(0,  act[0], act[1], act[2], act[3], 8);   // d1   @256
    pool(act[3], act[0], 8, 128);
    bottleneck(4,  act[0], act[1], act[2], act[3], 7);   // d1_1 @128
    bottleneck(8,  act[3], act[0], act[1], act[2], 7);   // d2   @128
    pool(act[2], act[0], 7, 256);
    bottleneck(12, act[0], act[1], act[2], act[3], 6);   // d2_1 @64
    bottleneck(16, act[3], act[0], act[1], act[2], 6);   // d3   @64
    pool(act[2], act[0], 6, 512);
    bottleneck(20, act[0], act[1], act[2], act[3], 5);   // d3_1 @32
    ConvP pf = {act[3], cv[24].wt, cv[24].b, nullptr, nullptr, nullptr,
                nullptr, (float*)d_out, 5, 1024, 0};
    launch_conv(3, 512, pf, stream);                      // d4 -> fp32 NCHW
}
